// CopyNetDecoder_15582141349896
// MI455X (gfx1250) — matrix operand order repaired and checked by reading, not compiled, not run on hardware
//
#include <hip/hip_runtime.h>
#include <hip/hip_bf16.h>
#include <math.h>

#define NEG (-1.0e10f)

typedef __attribute__((ext_vector_type(16))) __bf16 v16bf;
typedef __attribute__((ext_vector_type(8)))  __bf16 v8bf;
typedef __attribute__((ext_vector_type(8)))  float  v8f;

// ---------------------------------------------------------------------------
// enc_bf[(b*512+l)*1024+k] = (bf16) encoder_outputs[(l*32+b)*1024+k]
__global__ void conv_enc_kernel(const float* __restrict__ enc, __bf16* __restrict__ out) {
  size_t i = (size_t)blockIdx.x * blockDim.x + threadIdx.x;   // 16384*1024 elems
  int k = (int)(i & 1023);
  int m = (int)(i >> 10);
  int b = m >> 9, l = m & 511;
  out[i] = (__bf16)enc[((size_t)l * 32 + b) * 1024 + k];
}

// wa_bf[n*1024+k] = (bf16) W_attn[(1024+k)*1024 + n]   (enc block, packed [n][k])
__global__ void conv_wa_kernel(const float* __restrict__ W, __bf16* __restrict__ out) {
  size_t i = (size_t)blockIdx.x * blockDim.x + threadIdx.x;   // 1024*1024
  int n = (int)(i & 1023);
  int k = (int)(i >> 10);
  out[(size_t)n * 1024 + k] = (__bf16)W[(size_t)(1024 + k) * 1024 + n];
}

// straight cast (W_copy is already [n][k])
__global__ void conv_cast_kernel(const float* __restrict__ in, __bf16* __restrict__ out, int count) {
  size_t i = (size_t)blockIdx.x * blockDim.x + threadIdx.x;
  if (i < (size_t)count) out[i] = (__bf16)in[i];
}

// ---------------------------------------------------------------------------
// Fused GEMM (16384x1024 @ 1024x1024 bf16) -> tanh -> dot(v) -> reduce N.
// 512 threads = 16 waves. Block covers 32 rows (2 M-tiles); each wave covers
// 4 N-tiles (N range [wave*64, +64)). Each B fragment feeds two WMMAs (2x B
// reuse vs one M-tile) while per-wave accumulators stay at 2*4*8 = 64 VGPRs
// (the spill-free budget observed in round 1).
// mode 0: outScore[m] += sum_n tanh(C + rowBias[b][n]) * Vg[n]      (attention)
// mode 1: outScore[b*510+l-1] += sum_n tanh(C + biasN[n]) * Vb[b][n] (copy proj)
__global__ void __launch_bounds__(512, 1)
wmma_tanh_dot(const __bf16* __restrict__ A,   // [16384][1024]
              const __bf16* __restrict__ Wp,  // [n][k] 1024x1024
              const float* __restrict__ biasN,
              const float* __restrict__ rowBias,
              const float* __restrict__ Vg,
              const float* __restrict__ Vb,
              float* __restrict__ outScore,
              int mode) {
  const int K = 1024;
  int m0   = blockIdx.x * 32;          // 32 rows (2 M-tiles) per block
  int wave = threadIdx.x >> 5;         // 0..15 -> N range [wave*64, +64)
  int lane = threadIdx.x & 31;
  int half = (lane >= 16) ? 1 : 0;
  int l16  = lane & 15;
  int b    = m0 >> 9;                  // batch index (32 | 512 -> same b for both tiles)

  v8f acc[2][4] = {};

  const __bf16* Arow0 = A + (size_t)(m0 + l16) * K;
  const __bf16* Arow1 = A + (size_t)(m0 + 16 + l16) * K;

  for (int k0 = 0; k0 < K; k0 += 32) {
    // A fragments: lane<16 -> M=l16, K = k0+{0..7,16..23}; lane>=16 -> K = k0+{8..15,24..31}
    int ka = k0 + (half ? 8 : 0);
    v8bf a0lo = *(const v8bf*)(Arow0 + ka);
    v8bf a0hi = *(const v8bf*)(Arow0 + ka + 16);
    v8bf a1lo = *(const v8bf*)(Arow1 + ka);
    v8bf a1hi = *(const v8bf*)(Arow1 + ka + 16);
    v16bf af0, af1;
#pragma unroll
    for (int i = 0; i < 8; i++) {
      af0[i] = a0lo[i]; af0[8 + i] = a0hi[i];
      af1[i] = a1lo[i]; af1[8 + i] = a1hi[i];
    }
#pragma unroll
    for (int t = 0; t < 4; t++) {
      // B fragment: N = tile*16 + l16, K = k0 + half*8 + {0..7,16..23}
      int n = (wave * 4 + t) * 16 + l16;
      const __bf16* Brow = Wp + (size_t)n * K + k0 + (half ? 8 : 0);
      v8bf b_lo = *(const v8bf*)(Brow);
      v8bf b_hi = *(const v8bf*)(Brow + 16);
      v16bf bfrag;
#pragma unroll
      for (int i = 0; i < 8; i++) { bfrag[i] = b_lo[i]; bfrag[8 + i] = b_hi[i]; }
      acc[0][t] = __builtin_amdgcn_wmma_f32_16x16x32_bf16(
          false, af0, false, bfrag, (short)0, acc[0][t], false, false);
      acc[1][t] = __builtin_amdgcn_wmma_f32_16x16x32_bf16(
          false, af1, false, bfrag, (short)0, acc[1][t], false, false);
    }
  }

  // Epilogue: C layout -> VGPR j, lanes 0-15: M=j, N=l16; lanes 16-31: M=8+j
  float partial[2][8];
#pragma unroll
  for (int mt = 0; mt < 2; mt++)
#pragma unroll
    for (int j = 0; j < 8; j++) partial[mt][j] = 0.f;

#pragma unroll
  for (int t = 0; t < 4; t++) {
    int n = (wave * 4 + t) * 16 + l16;
    float bn = biasN   ? biasN[n]              : 0.f;
    float rb = rowBias ? rowBias[b * 1024 + n] : 0.f;
    float v  = Vg      ? Vg[n]                 : Vb[b * 1024 + n];
#pragma unroll
    for (int mt = 0; mt < 2; mt++)
#pragma unroll
      for (int j = 0; j < 8; j++)
        partial[mt][j] += tanhf(acc[mt][t][j] + bn + rb) * v;
  }
  // reduce over the 16 lanes of each half (the N dimension within a tile)
#pragma unroll
  for (int mt = 0; mt < 2; mt++)
#pragma unroll
    for (int j = 0; j < 8; j++) {
      float s = partial[mt][j];
      for (int off = 1; off < 16; off <<= 1) s += __shfl_xor(s, off, 32);
      partial[mt][j] = s;
    }
  if (l16 < 8) {
    int row = l16 + half * 8;
#pragma unroll
    for (int mt = 0; mt < 2; mt++) {
      int m = m0 + mt * 16 + row;
      float val = partial[mt][l16];
      if (mode == 0) {
        atomicAdd(&outScore[m], val);
      } else {
        int l = m & 511;
        if (l >= 1 && l <= 510) atomicAdd(&outScore[(m >> 9) * 510 + (l - 1)], val);
      }
    }
  }
}

// ---------------------------------------------------------------------------
// out[b][n] = sum_k A[b][k]*W + (A2? sum_k A2[b][k]*W2) + b1[n] + b2[n]
// W: kmajorN==0 -> row-major [n][K]; else [k][kmajorN] (k-major, stride kmajorN)
#define TK 64
__global__ void gemm_m32(const float* __restrict__ A, int K,
                         const float* __restrict__ W, int kmajorN,
                         const float* __restrict__ A2, int K2,
                         const float* __restrict__ W2,
                         const float* __restrict__ b1, const float* __restrict__ b2,
                         float* __restrict__ out, int N, int outStride) {
  __shared__ float sA[32 * TK];
  int n = blockIdx.x * blockDim.x + threadIdx.x;
  float acc[32];
#pragma unroll
  for (int b = 0; b < 32; b++) acc[b] = 0.f;

  for (int k0 = 0; k0 < K; k0 += TK) {
    __syncthreads();
    for (int e = threadIdx.x; e < 32 * TK; e += blockDim.x)
      sA[e] = A[(size_t)(e / TK) * K + k0 + (e % TK)];
    __syncthreads();
    if (n < N) {
      if (kmajorN) {
        for (int k = 0; k < TK; k++) {
          float w = W[(size_t)(k0 + k) * kmajorN + n];
#pragma unroll
          for (int b = 0; b < 32; b++) acc[b] += w * sA[b * TK + k];
        }
      } else {
        const float* Wr = W + (size_t)n * K + k0;
        if (k0 + TK < K) __builtin_prefetch(Wr + TK, 0, 3);   // global_prefetch next K-tile
        for (int k = 0; k < TK; k += 4) {
          float4 w4 = *(const float4*)(Wr + k);
#pragma unroll
          for (int b = 0; b < 32; b++) {
            acc[b] += w4.x * sA[b * TK + k]     + w4.y * sA[b * TK + k + 1]
                    + w4.z * sA[b * TK + k + 2] + w4.w * sA[b * TK + k + 3];
          }
        }
      }
    }
  }
  if (A2) {
    for (int k0 = 0; k0 < K2; k0 += TK) {
      __syncthreads();
      for (int e = threadIdx.x; e < 32 * TK; e += blockDim.x)
        sA[e] = A2[(size_t)(e / TK) * K2 + k0 + (e % TK)];
      __syncthreads();
      if (n < N) {
        const float* Wr = W2 + (size_t)n * K2 + k0;
        if (k0 + TK < K2) __builtin_prefetch(Wr + TK, 0, 3);
        for (int k = 0; k < TK; k += 4) {
          float4 w4 = *(const float4*)(Wr + k);
#pragma unroll
          for (int b = 0; b < 32; b++) {
            acc[b] += w4.x * sA[b * TK + k]     + w4.y * sA[b * TK + k + 1]
                    + w4.z * sA[b * TK + k + 2] + w4.w * sA[b * TK + k + 3];
          }
        }
      }
    }
  }
  if (n < N) {
    float bias = (b1 ? b1[n] : 0.f) + (b2 ? b2[n] : 0.f);
#pragma unroll
    for (int b = 0; b < 32; b++) out[(size_t)b * outStride + n] = acc[b] + bias;
  }
}

// ---------------------------------------------------------------------------
__global__ void softmax_attn(const float* __restrict__ scores, const int* __restrict__ mask,
                             float* __restrict__ attn) {
  __shared__ float sd[512];
  int b = blockIdx.x, l = threadIdx.x;
  float v = scores[b * 512 + l];
  if (mask[b * 512 + l] == 0) v = NEG;
  sd[l] = v; __syncthreads();
  for (int s = 256; s > 0; s >>= 1) { if (l < s) sd[l] = fmaxf(sd[l], sd[l + s]); __syncthreads(); }
  float mx = sd[0]; __syncthreads();
  float e = __expf(v - mx);
  sd[l] = e; __syncthreads();
  for (int s = 256; s > 0; s >>= 1) { if (l < s) sd[l] += sd[l + s]; __syncthreads(); }
  attn[b * 512 + l] = e / sd[0];
}

__global__ void sel_softmax(const int* __restrict__ src, const int* __restrict__ inp,
                            const int* __restrict__ s2c, const float* __restrict__ selw,
                            const int* __restrict__ mask, float* __restrict__ sel) {
  __shared__ float sd[512];
  int b = blockIdx.x, l = threadIdx.x;       // l < 510 active
  bool act = (l < 510);
  float v = NEG;
  if (act) {
    int conv = s2c[src[(l + 1) * 32 + b]];
    v = (conv == inp[b]) ? selw[b * 510 + l] : NEG;
  }
  sd[l] = v; __syncthreads();
  for (int s = 256; s > 0; s >>= 1) { if (l < s) sd[l] = fmaxf(sd[l], sd[l + s]); __syncthreads(); }
  float mx = sd[0]; __syncthreads();
  float e = act ? __expf(v - mx) : 0.f;
  sd[l] = e; __syncthreads();
  for (int s = 256; s > 0; s >>= 1) { if (l < s) sd[l] += sd[l + s]; __syncthreads(); }
  if (act) {
    float r = e / sd[0];
    if (mask[b * 512 + (l + 1)] == 0) r = NEG;
    sel[b * 510 + l] = r;
  }
}

// out[b][h] = sum_l w[b][l] * enc[(l+loff)][b][h]
__global__ void weighted_sum(const float* __restrict__ w, int Lcnt, int loff, int wstride,
                             const float* __restrict__ enc, float* __restrict__ out) {
  int b = blockIdx.y;
  int h = blockIdx.x * blockDim.x + threadIdx.x;   // < 1024
  float acc = 0.f;
  for (int l = 0; l < Lcnt; l++)
    acc += w[b * wstride + l] * enc[((size_t)(l + loff) * 32 + b) * 1024 + h];
  out[b * 1024 + h] = acc;
}

// x = [embedded(512), attentive_read(1024), selective_read(1024)]
__global__ void build_x(const float* __restrict__ emb, const int* __restrict__ inp,
                        const float* __restrict__ ar, const float* __restrict__ sr,
                        float* __restrict__ x) {
  int b = blockIdx.y;
  int i = blockIdx.x * blockDim.x + threadIdx.x;   // < 2560
  float v;
  if (i < 512)       v = emb[(size_t)inp[b] * 512 + i];
  else if (i < 1536) v = ar[b * 1024 + (i - 512)];
  else               v = sr[b * 1024 + (i - 1536)];
  x[b * 2560 + i] = v;
}

// x_gen = [h1(1024), attentive_read(1024), embedded(512)]
__global__ void build_xgen(const float* __restrict__ h1, const float* __restrict__ ar,
                           const float* __restrict__ emb, const int* __restrict__ inp,
                           float* __restrict__ x) {
  int b = blockIdx.y;
  int i = blockIdx.x * blockDim.x + threadIdx.x;   // < 2560
  float v;
  if (i < 1024)      v = h1[b * 1024 + i];
  else if (i < 2048) v = ar[b * 1024 + (i - 1024)];
  else               v = emb[(size_t)inp[b] * 512 + (i - 2048)];
  x[b * 2560 + i] = v;
}

__device__ __forceinline__ float sigf(float x) { return 1.f / (1.f + __expf(-x)); }

__global__ void lstm_act(const float* __restrict__ g, const float* __restrict__ c_old,
                         float* __restrict__ h_new, float* __restrict__ c_new) {
  int b = blockIdx.y;
  int j = blockIdx.x * blockDim.x + threadIdx.x;   // < 1024
  const float* gb = g + (size_t)b * 4096;
  float cn = sigf(gb[1024 + j]) * c_old[b * 1024 + j] + sigf(gb[j]) * tanhf(gb[2048 + j]);
  float hn = sigf(gb[3072 + j]) * tanhf(cn);
  c_new[b * 1024 + j] = cn;
  h_new[b * 1024 + j] = hn;
}

__global__ void copy_final(const float* __restrict__ raw, const int* __restrict__ mask,
                           float* __restrict__ cpy) {
  int b = blockIdx.y;
  int l = blockIdx.x * blockDim.x + threadIdx.x;
  if (l < 510)
    cpy[b * 510 + l] = (mask[b * 512 + l + 1] == 0) ? NEG : raw[b * 510 + l];
}

__global__ void scatter_copy(const float* __restrict__ cpy, const int* __restrict__ src,
                             const int* __restrict__ s2c, float* __restrict__ preds) {
  int b = blockIdx.y;
  int l = blockIdx.x * blockDim.x + threadIdx.x;
  if (l < 510) {
    int idx = s2c[src[(l + 1) * 32 + b]];
    preds[(size_t)b * 60000 + idx] = cpy[b * 510 + l];
  }
}

// ---------------------------------------------------------------------------
extern "C" void kernel_launch(void* const* d_in, const int* in_sizes, int n_in,
                              void* d_out_, int out_size, void* d_ws, size_t ws_size,
                              hipStream_t stream) {
  (void)in_sizes; (void)n_in; (void)out_size; (void)ws_size;
  const int*   src    = (const int*)d_in[0];
  const int*   inp    = (const int*)d_in[1];
  const float* hidden = (const float*)d_in[2];
  const float* c_in   = (const float*)d_in[3];
  const float* enc    = (const float*)d_in[4];
  const int*   mask   = (const int*)d_in[5];
  const float* selw   = (const float*)d_in[6];
  const int*   s2c    = (const int*)d_in[7];
  const float* emb    = (const float*)d_in[8];
  const float* W_attn = (const float*)d_in[9];
  const float* b_attn = (const float*)d_in[10];
  const float* v_attn = (const float*)d_in[11];
  const float* W_ih0  = (const float*)d_in[12];
  const float* W_hh0  = (const float*)d_in[13];
  const float* b_ih0  = (const float*)d_in[14];
  const float* b_hh0  = (const float*)d_in[15];
  const float* W_ih1  = (const float*)d_in[16];
  const float* W_hh1  = (const float*)d_in[17];
  const float* b_ih1  = (const float*)d_in[18];
  const float* b_hh1  = (const float*)d_in[19];
  const float* W_fc   = (const float*)d_in[20];
  const float* b_fc   = (const float*)d_in[21];
  const float* W_copy = (const float*)d_in[22];
  const float* b_copy = (const float*)d_in[23];

  float* out    = (float*)d_out_;
  float* preds  = out;                 // 32*60000
  float* h_new  = out + 1920000;       // (2,32,1024)
  float* c_new  = out + 1985536;       // (2,32,1024)
  float* attn_w = out + 2051072;       // (32,512)
  float* cpyout = out + 2067456;       // (32,510)

  char* ws = (char*)d_ws;
  __bf16* enc_bf = (__bf16*)(ws + 0);           // 33,554,432 B
  __bf16* wa_bf  = (__bf16*)(ws + 33554432);    //  2,097,152 B
  __bf16* wc_bf  = (__bf16*)(ws + 35651584);    //  2,097,152 B
  float* hpart   = (float*)(ws + 37748736);     //    131,072 B
  float* scores  = (float*)(ws + 37879808);     //     65,536 B
  float* sel     = (float*)(ws + 37945344);     //     65,536 B
  float* ar      = (float*)(ws + 38010880);     //    131,072 B
  float* sr      = (float*)(ws + 38141952);     //    131,072 B
  float* xbuf    = (float*)(ws + 38273024);     //    327,680 B
  float* gbuf    = (float*)(ws + 38600704);     //    524,288 B
  float* xgen    = (float*)(ws + 39124992);     //    327,680 B
  float* raw     = (float*)(ws + 39452672);     //     65,536 B

  hipMemsetAsync(preds, 0, (size_t)1920000 * 4, stream);
  hipMemsetAsync(scores, 0, (size_t)16384 * 4, stream);
  hipMemsetAsync(raw, 0, (size_t)32 * 510 * 4, stream);

  conv_enc_kernel<<<65536, 256, 0, stream>>>(enc, enc_bf);
  conv_wa_kernel<<<4096, 256, 0, stream>>>(W_attn, wa_bf);
  conv_cast_kernel<<<4096, 256, 0, stream>>>(W_copy, wc_bf, 1024 * 1024);

  // hpart = h_top @ W_attn[:1024] + b_attn  (W_attn top block is k-major)
  gemm_m32<<<4, 256, 0, stream>>>(hidden + 32768, 1024, W_attn, 1024,
                                  nullptr, 0, nullptr, b_attn, nullptr,
                                  hpart, 1024, 1024);

  // scores[m] = sum_n tanh(enc@Wa + hpart) * v_attn   (34 GFLOP, bf16 WMMA)
  wmma_tanh_dot<<<512, 512, 0, stream>>>(enc_bf, wa_bf, nullptr, hpart,
                                         v_attn, nullptr, scores, 0);

  softmax_attn<<<32, 512, 0, stream>>>(scores, mask, attn_w);
  sel_softmax<<<32, 512, 0, stream>>>(src, inp, s2c, selw, mask, sel);

  weighted_sum<<<dim3(4, 32), 256, 0, stream>>>(attn_w, 512, 0, 512, enc, ar);
  weighted_sum<<<dim3(4, 32), 256, 0, stream>>>(sel, 510, 1, 510, enc, sr);

  build_x<<<dim3(10, 32), 256, 0, stream>>>(emb, inp, ar, sr, xbuf);

  // layer 0
  gemm_m32<<<16, 256, 0, stream>>>(xbuf, 2560, W_ih0, 0,
                                   hidden, 1024, W_hh0, b_ih0, b_hh0,
                                   gbuf, 4096, 4096);
  lstm_act<<<dim3(4, 32), 256, 0, stream>>>(gbuf, c_in, h_new, c_new);

  // layer 1
  gemm_m32<<<16, 256, 0, stream>>>(h_new, 1024, W_ih1, 0,
                                   hidden + 32768, 1024, W_hh1, b_ih1, b_hh1,
                                   gbuf, 4096, 4096);
  lstm_act<<<dim3(4, 32), 256, 0, stream>>>(gbuf, c_in + 32768,
                                            h_new + 32768, c_new + 32768);

  build_xgen<<<dim3(10, 32), 256, 0, stream>>>(h_new + 32768, ar, emb, inp, xgen);

  // gen = x_gen @ W_fc^T + b_fc  -> preds[:, :32000]   (327 MB, memory-bound f32)
  gemm_m32<<<125, 256, 0, stream>>>(xgen, 2560, W_fc, 0,
                                    nullptr, 0, nullptr, b_fc, nullptr,
                                    preds, 32000, 60000);

  // copy_scores_raw = sum_n tanh(enc@Wc^T + b_copy) * h1   (34 GFLOP, bf16 WMMA)
  wmma_tanh_dot<<<512, 512, 0, stream>>>(enc_bf, wc_bf, b_copy, nullptr,
                                         nullptr, h_new + 32768, raw, 1);

  copy_final<<<dim3(2, 32), 256, 0, stream>>>(raw, mask, cpyout);
  scatter_copy<<<dim3(2, 32), 256, 0, stream>>>(cpyout, src, s2c, preds);
}
